// GatingAttention_28200755266186
// MI455X (gfx1250) — matrix operand order repaired and checked by reading, not compiled, not run on hardware
//
#include <hip/hip_runtime.h>

#define H_ 16
#define S_ 512
#define F_ 512
#define D_ 64
#define B_ 8
#define RANK_ 12
#define TOPK_ 51
#define LN_EPS_ 1e-5f
#define ALPHA_SCALE_ 0.04419417382415922f  /* 1/sqrt(512) */
#define VPAD_ 8                            /* 16-byte LDS row padding (halfs) */

typedef __attribute__((ext_vector_type(16))) _Float16 v16h;
typedef __attribute__((ext_vector_type(2)))  _Float16 v2h;
typedef __attribute__((ext_vector_type(8)))  float    v8f;
typedef __attribute__((ext_vector_type(2)))  float    v2f;
typedef __attribute__((ext_vector_type(4)))  unsigned v4u;
typedef __attribute__((ext_vector_type(8)))  int      v8i;
typedef __attribute__((ext_vector_type(4)))  int      v4i;

// ---------------- wave32 helpers ----------------
__device__ __forceinline__ float wred_add(float v) {
  #pragma unroll
  for (int m = 16; m >= 1; m >>= 1) v += __shfl_xor(v, m, 32);
  return v;
}
__device__ __forceinline__ float wred_max(float v) {
  #pragma unroll
  for (int m = 16; m >= 1; m >>= 1) v = fmaxf(v, __shfl_xor(v, m, 32));
  return v;
}
__device__ __forceinline__ int wred_addi(int v) {
  #pragma unroll
  for (int m = 16; m >= 1; m >>= 1) v += __shfl_xor(v, m, 32);
  return v;
}
// monotone float -> u32 key (total order matching float order)
__device__ __forceinline__ unsigned mono_key(float x) {
  unsigned u = __float_as_uint(x);
  return (u & 0x80000000u) ? ~u : (u | 0x80000000u);
}

// Per-wave: row of 512 logits held 16/lane. Exact top-k softmax via
// 32-round radix search for the k-th largest key (wave-uniform counts).
__device__ __forceinline__ void topk_softmax16(const float* x, float* p) {
  unsigned key[16];
  #pragma unroll
  for (int i = 0; i < 16; ++i) key[i] = mono_key(x[i]);
  unsigned kth = 0u;
  #pragma unroll 1
  for (int bit = 31; bit >= 0; --bit) {
    unsigned trial = kth | (1u << bit);
    int c = 0;
    #pragma unroll
    for (int i = 0; i < 16; ++i) c += (key[i] >= trial) ? 1 : 0;
    c = wred_addi(c);
    if (c >= TOPK_) kth = trial;
  }
  float mx = -3.4e38f;
  #pragma unroll
  for (int i = 0; i < 16; ++i) mx = fmaxf(mx, x[i]);
  mx = wred_max(mx);
  float s = 0.f;
  #pragma unroll
  for (int i = 0; i < 16; ++i) {
    float e = (key[i] >= kth) ? __expf(x[i] - mx) : 0.f;
    p[i] = e;
    s += e;
  }
  s = wred_add(s);
  float inv = 1.f / s;
  #pragma unroll
  for (int i = 0; i < 16; ++i) p[i] *= inv;
}

// ---------------- kernel A: energy -> RMS -> gain -> LayerNorm ----------------
__global__ __launch_bounds__(256)
void score_kernel(const float* __restrict__ values, const float* __restrict__ temp,
                  const float* __restrict__ ln_w, const float* __restrict__ ln_b,
                  float* __restrict__ score) {
  __shared__ float eng[F_];
  __shared__ float red[256];
  const int b = blockIdx.x / H_;
  const int h = blockIdx.x % H_;
  const int t = threadIdx.x;

  for (int f = t; f < F_; f += 256) {
    const float* vp = values + (((size_t)b * F_ + f) * H_ + h) * D_;
    float acc = 0.f;
    #pragma unroll
    for (int d = 0; d < D_; d += 4) {
      float4 v = *(const float4*)(vp + d);
      acc += v.x * v.x + v.y * v.y + v.z * v.z + v.w * v.w;
    }
    eng[f] = acc * (1.0f / D_);
  }
  __syncthreads();

  float e0 = eng[t], e1 = eng[t + 256];
  red[t] = e0 + e1;
  __syncthreads();
  #pragma unroll 1
  for (int o = 128; o > 0; o >>= 1) { if (t < o) red[t] += red[t + o]; __syncthreads(); }
  const float rms = fmaxf(sqrtf(red[0] * (1.0f / F_)), 1e-6f);
  __syncthreads();

  const float tv = temp[h];
  const float gain = (tv > 20.f) ? tv : log1pf(__expf(tv));   // softplus
  float s0 = e0 / rms * gain;
  float s1 = e1 / rms * gain;

  red[t] = s0 + s1;
  __syncthreads();
  #pragma unroll 1
  for (int o = 128; o > 0; o >>= 1) { if (t < o) red[t] += red[t + o]; __syncthreads(); }
  const float mu = red[0] * (1.0f / F_);
  __syncthreads();

  red[t] = (s0 - mu) * (s0 - mu) + (s1 - mu) * (s1 - mu);
  __syncthreads();
  #pragma unroll 1
  for (int o = 128; o > 0; o >>= 1) { if (t < o) red[t] += red[t + o]; __syncthreads(); }
  const float var = red[0] * (1.0f / F_);
  const float inv = rsqrtf(var + LN_EPS_);

  float* op = score + ((size_t)(b * H_ + h)) * F_;
  op[t]       = (s0 - mu) * inv * ln_w[t]       + ln_b[t];
  op[t + 256] = (s1 - mu) * inv * ln_w[t + 256] + ln_b[t + 256];
}

// ---------------- kernel B: bilinear = U @ V via WMMA f32 16x16x4 ----------------
__global__ __launch_bounds__(256)
void bilinear_kernel(const float* __restrict__ U, const float* __restrict__ V,
                     float* __restrict__ bil) {
  const int wave = threadIdx.x >> 5;
  const int lane = threadIdx.x & 31;
  const int tile = blockIdx.x * 8 + wave;     // h*(S/16)*(F/16) tiles
  const int ft = tile & 31;
  const int st = (tile >> 5) & 31;
  const int h  = tile >> 10;
  const int s0 = st * 16, f0 = ft * 16;
  const int Lm = lane & 15;
  const bool hi = lane >= 16;

  v8f c = {};
#if __has_builtin(__builtin_amdgcn_wmma_f32_16x16x4_f32)
  #pragma unroll
  for (int k0 = 0; k0 < RANK_; k0 += 4) {
    const int ra = k0 + (hi ? 2 : 0);
    v2f a, bf;
    a.x  = U[((size_t)(h * S_ + s0 + Lm)) * RANK_ + ra];
    a.y  = U[((size_t)(h * S_ + s0 + Lm)) * RANK_ + ra + 1];
    bf.x = V[((size_t)(h * RANK_ + ra))     * F_ + f0 + Lm];
    bf.y = V[((size_t)(h * RANK_ + ra + 1)) * F_ + f0 + Lm];
    c = __builtin_amdgcn_wmma_f32_16x16x4_f32(false, a, false, bf, (short)0, c,
                                              false, false);
  }
#else
  #pragma unroll
  for (int i = 0; i < 8; ++i) {
    const int s = s0 + i + (hi ? 8 : 0);
    const int f = f0 + Lm;
    float acc = 0.f;
    #pragma unroll
    for (int r = 0; r < RANK_; ++r)
      acc += U[(size_t)(h * S_ + s) * RANK_ + r] * V[(size_t)(h * RANK_ + r) * F_ + f];
    c[i] = acc;
  }
#endif
  #pragma unroll
  for (int i = 0; i < 8; ++i) {
    const int s = s0 + i + (hi ? 8 : 0);
    bil[((size_t)(h * S_ + s)) * F_ + f0 + Lm] = c[i];
  }
}

// ---------------- kernel C: attn_alpha = topk-softmax(alpha / sqrt(F)) ----------------
__global__ __launch_bounds__(256)
void alpha_attn_kernel(const float* __restrict__ alpha, float* __restrict__ attnA) {
  const int wave = threadIdx.x >> 5;
  const int lane = threadIdx.x & 31;
  const int row  = blockIdx.x * 8 + wave;     // row in [0, H*S)
  const float* ap = alpha + (size_t)row * F_ + lane * 16;
  float x[16];
  #pragma unroll
  for (int i = 0; i < 16; i += 4) {
    float4 v = *(const float4*)(ap + i);
    x[i + 0] = v.x * ALPHA_SCALE_; x[i + 1] = v.y * ALPHA_SCALE_;
    x[i + 2] = v.z * ALPHA_SCALE_; x[i + 3] = v.w * ALPHA_SCALE_;
  }
  float p[16];
  topk_softmax16(x, p);
  float* op = attnA + (size_t)row * F_ + lane * 16;
  #pragma unroll
  for (int i = 0; i < 16; i += 4) {
    float4 v; v.x = p[i]; v.y = p[i + 1]; v.z = p[i + 2]; v.w = p[i + 3];
    *(float4*)(op + i) = v;
  }
}

// ---------------- kernel E: pre-transpose values to f16 [B,H,D,F] ----------------
__global__ __launch_bounds__(256)
void values_f16_kernel(const float* __restrict__ values, _Float16* __restrict__ vals16) {
  __shared__ _Float16 tile[64][64 + 4];
  const int b = blockIdx.x / H_;
  const int h = blockIdx.x % H_;
  const int t = threadIdx.x;
  #pragma unroll 1
  for (int fc = 0; fc < F_ / 64; ++fc) {
    const int f0 = fc * 64;
    #pragma unroll 1
    for (int k = 0; k < 16; ++k) {        // load: lanes -> consecutive d (coalesced)
      const int idx = t + k * 256;
      const int fl = idx >> 6, d = idx & 63;
      tile[d][fl] = (_Float16)values[(((size_t)b * F_ + f0 + fl) * H_ + h) * D_ + d];
    }
    __syncthreads();
    #pragma unroll 1
    for (int k = 0; k < 16; ++k) {        // store: lanes -> consecutive f (coalesced)
      const int idx = t + k * 256;
      const int d = idx >> 6, fl = idx & 63;
      vals16[((size_t)(b * H_ + h) * D_ + d) * F_ + f0 + fl] = tile[d][fl];
    }
    __syncthreads();
  }
}

// ---------------- TDM: one-descriptor DMA of the [64 d][512 f] f16 tile -------------
// D# pads LDS dest by 16B every 1024B row -> lands in vals_t[64][F_+VPAD_] layout.
__device__ __forceinline__ void tdm_load_tile(unsigned lds_addr, const _Float16* gptr) {
#if __has_builtin(__builtin_amdgcn_tensor_load_to_lds)
  const unsigned long long ga = (unsigned long long)gptr;
  v4u g0;
  g0[0] = 1u;                                            // count=1 (valid), user mode
  g0[1] = lds_addr;                                      // LDS byte address
  g0[2] = (unsigned)ga;                                  // global_addr[31:0]
  g0[3] = (unsigned)((ga >> 32) & 0x01FFFFFFu)
        | 0x80000000u;                                   // global_addr[56:32] | type=2
  v8i g1;
  g1[0] = (int)((1u << 16)      // data_size = 1 (2 bytes)
              | (1u << 20)      // pad_enable
              | (7u << 22)      // pad_interval: every 128*8B = 1024B
              | (3u << 25));    // pad_amount: 4 DWORDs = 16B
  g1[1] = (int)(512u << 16);    // tensor_dim0 = 512 (bits 79:48)
  g1[2] = (int)(64u << 16);     // tensor_dim1 = 64  (bits 111:80)
  g1[3] = (int)(512u << 16);    // tile_dim0 = 512   (bits 127:112)
  g1[4] = 64;                   // tile_dim1 = 64, tile_dim2 = 0
  g1[5] = 512;                  // tensor_dim0_stride = 512 elements
  g1[6] = 0;
  g1[7] = 0;
  v4i gz = {};
#if __has_include(<hip/amd_detail/amd_gfx1250_TDM.h>)
  v8i gz8 = {};
  __builtin_amdgcn_tensor_load_to_lds(g0, g1, gz, gz, gz8, 0);
#else
  __builtin_amdgcn_tensor_load_to_lds(g0, g1, gz, gz, 0);
#endif
#endif
}
__device__ __forceinline__ void wait_tensorcnt0() {
#if __has_builtin(__builtin_amdgcn_s_wait_tensorcnt)
  __builtin_amdgcn_s_wait_tensorcnt(0);
#else
  asm volatile("s_wait_tensorcnt 0x0" ::: "memory");
#endif
}

// ---------------- kernel D: fused data-attn + einsum via WMMA f16 ----------------
__global__ __launch_bounds__(512)
void fused_attn_out_kernel(const _Float16* __restrict__ vals16, const float* __restrict__ gamma,
                           const float* __restrict__ score, const float* __restrict__ bil,
                           const float* __restrict__ attnA, float* __restrict__ out) {
  __shared__ _Float16 vals_t[D_][F_ + VPAD_];   // [d][f], filled by TDM (padded rows)
  __shared__ _Float16 att[64][F_ + VPAD_];      // [s-row][f]

  const int blk = blockIdx.x;                   // b*H*8 + h*8 + st
  const int st = blk & 7;
  const int h  = (blk >> 3) & 15;
  const int b  = blk >> 7;
  const int s0 = st * 64;
  const int t  = threadIdx.x;
  const int wave = t >> 5, lane = t & 31;

  const _Float16* gsrc = vals16 + (size_t)(b * H_ + h) * D_ * F_;

#if __has_builtin(__builtin_amdgcn_tensor_load_to_lds)
  // phase 1: wave 0 launches the async TDM tile load; overlaps with phase 2
  if (wave == 0) {
    tdm_load_tile((unsigned)(unsigned long long)&vals_t[0][0], gsrc);
  }
#else
  for (int i = t; i < D_ * F_; i += 512) {
    const int d = i >> 9, f = i & (F_ - 1);
    vals_t[d][f] = gsrc[i];
  }
#endif

  // phase 2: each wave produces 4 attn_mix rows (f16 into LDS)
  #pragma unroll 1
  for (int rr = 0; rr < 4; ++rr) {
    const int srow = wave * 4 + rr;
    const int s = s0 + srow;
    const float* bp = bil   + ((size_t)(h * S_ + s)) * F_ + lane * 16;
    const float* sp = score + ((size_t)(b * H_ + h)) * F_ + lane * 16;
    const float g = gamma[h * S_ + s];
    float x[16];
    #pragma unroll
    for (int i = 0; i < 16; i += 4) {
      float4 bv = *(const float4*)(bp + i);
      float4 sv = *(const float4*)(sp + i);
      x[i + 0] = sv.x + bv.x + g; x[i + 1] = sv.y + bv.y + g;
      x[i + 2] = sv.z + bv.z + g; x[i + 3] = sv.w + bv.w + g;
    }
    float p[16];
    topk_softmax16(x, p);
    const float* apn = attnA + ((size_t)(h * S_ + s)) * F_ + lane * 16;
    #pragma unroll
    for (int i = 0; i < 16; i += 4) {
      float4 av = *(const float4*)(apn + i);
      v2h pk0; pk0.x = (_Float16)(p[i + 0] + av.x); pk0.y = (_Float16)(p[i + 1] + av.y);
      v2h pk1; pk1.x = (_Float16)(p[i + 2] + av.z); pk1.y = (_Float16)(p[i + 3] + av.w);
      *(v2h*)&att[srow][lane * 16 + i]     = pk0;
      *(v2h*)&att[srow][lane * 16 + i + 2] = pk1;
    }
  }

#if __has_builtin(__builtin_amdgcn_tensor_load_to_lds)
  if (wave == 0) wait_tensorcnt0();
#endif
  __syncthreads();

  // phase 3: 16 waves -> 4x4 grid of 16x16 C tiles, K=512 in chunks of 32
  const int mt = wave >> 2, nt = wave & 3;
  const int m0 = mt * 16, n0 = nt * 16;
  const int Lm = lane & 15;
  const bool hiL = lane >= 16;

  v8f c = {};
  #pragma unroll 1
  for (int f0 = 0; f0 < F_; f0 += 32) {
    v16h a, bm;
    // A fragment: 16x32, M = lane%16; lanes<16: K {0..7,16..23}; lanes>=16: K {8..15,24..31}
    const int arow = m0 + Lm;
    const int ka = hiL ? 8 : 0;
    #pragma unroll
    for (int j = 0; j < 4; ++j) {
      v2h t0 = *(const v2h*)&att[arow][f0 + ka + 2 * j];
      a[2 * j]     = t0.x; a[2 * j + 1]     = t0.y;
      v2h t1 = *(const v2h*)&att[arow][f0 + 16 + ka + 2 * j];
      a[8 + 2 * j] = t1.x; a[8 + 2 * j + 1] = t1.y;
    }
    // B fragment: 32x16, N = lane%16; same K order as A: lanes<16: K {0..7,16..23}; lanes>=16: K {8..15,24..31}
    const int nn = n0 + Lm;
    const int kb = hiL ? 8 : 0;
    #pragma unroll
    for (int j = 0; j < 8; ++j) {
      v2h t2 = *(const v2h*)&vals_t[nn][f0 + kb + 2 * j + (j >= 4 ? 8 : 0)];
      bm[2 * j] = t2.x; bm[2 * j + 1] = t2.y;
    }
    c = __builtin_amdgcn_wmma_f32_16x16x32_f16(false, a, false, bm, (short)0, c,
                                               false, false);
  }

  // C layout: VGPR i -> rows (m0+i | m0+8+i), N = lane%16
  #pragma unroll
  for (int i = 0; i < 8; ++i) {
    const int s = s0 + m0 + i + (hiL ? 8 : 0);
    const int d = n0 + Lm;
    out[(((size_t)b * S_ + s) * H_ + h) * D_ + d] = c[i];
  }
}

// ---------------- launch ----------------
extern "C" void kernel_launch(void* const* d_in, const int* in_sizes, int n_in,
                              void* d_out, int out_size, void* d_ws, size_t ws_size,
                              hipStream_t stream) {
  const float* values = (const float*)d_in[0];
  const float* alpha  = (const float*)d_in[1];
  const float* temp   = (const float*)d_in[2];
  const float* gamma  = (const float*)d_in[3];
  const float* U      = (const float*)d_in[4];
  const float* V      = (const float*)d_in[5];
  const float* ln_w   = (const float*)d_in[6];
  const float* ln_b   = (const float*)d_in[7];
  float* out = (float*)d_out;

  float* ws    = (float*)d_ws;
  float* score = ws;                                   // B*H*F   = 65536 f32
  float* bil   = score + (size_t)B_ * H_ * F_;         // H*S*F   = 4194304 f32
  float* attnA = bil + (size_t)H_ * S_ * F_;           // H*S*F   = 4194304 f32
  _Float16* vals16 = (_Float16*)(attnA + (size_t)H_ * S_ * F_);  // B*H*D*F f16 = 8MB

  score_kernel<<<B_ * H_, 256, 0, stream>>>(values, temp, ln_w, ln_b, score);
  bilinear_kernel<<<(H_ * (S_ / 16) * (F_ / 16)) / 8, 256, 0, stream>>>(U, V, bil);
  alpha_attn_kernel<<<(H_ * S_) / 8, 256, 0, stream>>>(alpha, attnA);
  values_f16_kernel<<<B_ * H_, 256, 0, stream>>>(values, vals16);
  fused_attn_out_kernel<<<B_ * H_ * (S_ / 64), 512, 0, stream>>>(vals16, gamma, score,
                                                                 bil, attnA, out);
}
